// SelfTrans_52424370815568
// MI455X (gfx1250) — compile-verified, matrix operand order checked by reading, not run on hardware
//
#include <hip/hip_runtime.h>
#include <hip/hip_bf16.h>
#include <math.h>

typedef __attribute__((ext_vector_type(16))) __bf16 v16bf;
typedef __attribute__((ext_vector_type(4)))  __bf16 v4bf;
typedef __attribute__((ext_vector_type(8)))  float  v8f;

#define WMMA_BF16(a, b, c) \
  __builtin_amdgcn_wmma_f32_16x16x32_bf16(false, (a), false, (b), (short)0, (c), false, false)

enum {
  EPI_NONE = 0,
  EPI_BIAS = 1,
  EPI_BN = 2,
  EPI_BN_RELU = 3,
  EPI_BN_SIGMOID = 4,
  EPI_ADDPREV = 5,
  EPI_ADDPREV_LEAKY = 6,
};

// Build a WMMA 16-element fragment from row-contiguous bf16 data:
// element e -> p[khalf + (e&7) + ((e&8)?16:0)]   (two contiguous 8-runs)
static __device__ inline v16bf frag16(const __bf16* p, int khalf) {
  v16bf r;
  const v4bf a0 = *(const v4bf*)(p + khalf);
  const v4bf a1 = *(const v4bf*)(p + khalf + 4);
  const v4bf a2 = *(const v4bf*)(p + 16 + khalf);
  const v4bf a3 = *(const v4bf*)(p + 16 + khalf + 4);
  r[0] = a0[0];  r[1] = a0[1];  r[2] = a0[2];  r[3] = a0[3];
  r[4] = a1[0];  r[5] = a1[1];  r[6] = a1[2];  r[7] = a1[3];
  r[8] = a2[0];  r[9] = a2[1];  r[10] = a2[2]; r[11] = a2[3];
  r[12] = a3[0]; r[13] = a3[1]; r[14] = a3[2]; r[15] = a3[3];
  return r;
}

// Repack OIHW fp32 weights into bf16 WMMA A-fragment layout:
// dst[((tap*KC + kc)*CoT + ct)*512 + lane*16 + e]
// (kc-major over ct so one wave's MT consecutive tiles are contiguous)
__global__ void repack_w(const float* __restrict__ w, __bf16* __restrict__ d,
                         int Cout, int Cin, int KHW) {
  const long long total = (long long)KHW * Cout * Cin;
  const long long idx = (long long)blockIdx.x * blockDim.x + threadIdx.x;
  if (idx >= total) return;
  const int e = (int)(idx & 15);
  const int lane = (int)((idx >> 4) & 31);
  long long rest = idx >> 9;
  const int CoT = Cout >> 4;
  const int ct = (int)(rest % CoT);
  rest /= CoT;
  const int KC = Cin >> 5;
  const int kc = (int)(rest % KC);
  const int tap = (int)(rest / KC);
  const int row = ct * 16 + (lane & 15);
  const int K = (e & 7) + ((e & 8) ? 16 : 0) + ((lane & 16) ? 8 : 0);
  const int ci = kc * 32 + K;
  d[idx] = (__bf16)w[((size_t)row * Cin + ci) * KHW + tap];
}

// Fill padded columns [N, Npad) of the transposed q/k buffer:
// layout [b][g(q=0,k=1)][h][pos][64]; k row at pos==N = extra_k, rest zero.
__global__ void pad_qkT(__bf16* __restrict__ q, const float* __restrict__ ek,
                        int Bn, int N, int Npad) {
  const int PADW = Npad - N;
  const long long total = (long long)Bn * 2 * 4 * PADW * 64;
  const long long idx = (long long)blockIdx.x * blockDim.x + threadIdx.x;
  if (idx >= total) return;
  const int d = (int)(idx & 63);
  long long r1 = idx >> 6;
  const int j = (int)(r1 % PADW);
  r1 /= PADW;
  const int h = (int)(r1 & 3);
  const int g = (int)((r1 >> 2) & 1);
  const int b = (int)(r1 >> 3);
  const float val = (g == 1 && j == 0) ? ek[h * 64 + d] : 0.f;
  q[((((size_t)b * 2 + g) * 4 + h) * Npad + N + j) * 64 + d] = (__bf16)val;
}

// Fill padded columns of v ([ch][pos] layout): pos==N = extra_v, rest zero.
__global__ void pad_v(__bf16* __restrict__ v, const float* __restrict__ ev,
                      int Bn, int N, int Npad) {
  const int PADW = Npad - N;
  const long long total = (long long)Bn * 256 * PADW;
  const long long idx = (long long)blockIdx.x * blockDim.x + threadIdx.x;
  if (idx >= total) return;
  const int j = (int)(idx % PADW);
  const int ch = (int)((idx / PADW) % 256);
  const int b = (int)(idx / ((long long)PADW * 256));
  v[((size_t)b * 256 + ch) * Npad + N + j] = (__bf16)((j == 0) ? ev[ch] : 0.f);
}

// WMMA conv-as-GEMM with register blocking. Block = WAVES waves on one
// (batch, 16-position tile); wave w owns MT consecutive Cout tiles, sharing
// the single LDS-resident activation fragment across MT WMMAs per K-chunk.
// With WAVES*MT == CoT one block covers all of Cout, so each activation
// chunk is staged exactly once.
template <int KH, int KW, int PAD, int EPI, bool CONCAT, int WAVES, int MT,
          bool OBF, bool OPM>
__global__ __launch_bounds__(WAVES * 32) void conv_wmma(
    const float* __restrict__ src1, const float* __restrict__ src2,
    const __bf16* __restrict__ wp, const float* __restrict__ bias,
    const float* __restrict__ bng, const float* __restrict__ bnb,
    const float* __restrict__ bnm, const float* __restrict__ bnv,
    float* __restrict__ dstf, __bf16* __restrict__ dstb,
    int Bn, int Cin, int Cout, int H, int Wd, int Nd) {
  (void)Bn;
  const int N = H * Wd;
  const int NT = (N + 15) >> 4;
  const int CoT = Cout >> 4;
  const int KC = Cin >> 5;
  const int CG = CoT / (WAVES * MT);
  int bid = blockIdx.x;
  const int nt = bid % NT;
  bid /= NT;
  const int cg = bid % CG;
  const int b = bid / CG;

  const int tid = threadIdx.x;
  const int lane = tid & 31;
  const int wv = tid >> 5;
  const int ctb = (cg * WAVES + wv) * MT; // first of this wave's MT tiles
  const int m = lane & 15;
  const int khalf = (lane & 16) ? 8 : 0;
  const int ncol = (nt << 4) + m;
  const bool nvalid = ncol < N;
  const int Ch = CONCAT ? (Cin >> 1) : Cin;

  __shared__ __align__(32) __bf16 ldsb[512];
  __shared__ int sposArr[16];
  __shared__ float maskArr[16];

  const int THREADS = WAVES * 32;
  const int EPT = 512 / THREADS;

  v8f acc[MT];
  const v8f vzero = {};
#pragma unroll
  for (int j = 0; j < MT; ++j) acc[j] = vzero;

  for (int ky = 0; ky < KH; ++ky) {
    for (int kx = 0; kx < KW; ++kx) {
      const int tap = ky * KW + kx;
      if (tid < 16) {
        const int nc = (nt << 4) + tid;
        const bool v0 = nc < N;
        const int yy = (v0 ? nc / Wd : 0) + ky - PAD;
        const int xx = (v0 ? nc % Wd : 0) + kx - PAD;
        const bool ib = v0 && yy >= 0 && yy < H && xx >= 0 && xx < Wd;
        sposArr[tid] = ib ? (yy * Wd + xx) : 0;
        maskArr[tid] = ib ? 1.f : 0.f;
      }
      __syncthreads();

      for (int ci0 = 0; ci0 < Cin; ci0 += 32) {
        const float* sp;
        if (!CONCAT)
          sp = src1 + (size_t)b * Cin * N;
        else if (ci0 < Ch)
          sp = src1 + (size_t)b * Ch * N;
        else
          sp = src2 + ((size_t)b * Ch - (size_t)Ch) * N;

#pragma unroll
        for (int i = 0; i < EPT; ++i) {
          const int u = tid + i * THREADS;
          const int K = u >> 4;
          const int col = u & 15;
          const size_t ga = (size_t)(ci0 + K) * N + sposArr[col];
          const float xv = sp[ga] * maskArr[col];
          if (ci0 + 32 < Cin) __builtin_prefetch(sp + ga + (size_t)32 * N, 0, 3);
          const int dl = col | ((K & 8) ? 16 : 0);
          const int de = (K & 7) | ((K & 16) ? 8 : 0);
          ldsb[dl * 16 + de] = (__bf16)xv;
        }
        __syncthreads();

        const v16bf bv = *(const v16bf*)&ldsb[lane * 16];
        const __bf16* wpc =
            wp + (((size_t)tap * KC + (ci0 >> 5)) * CoT + ctb) * 512 + lane * 16;
#pragma unroll
        for (int j = 0; j < MT; ++j) {
          const v16bf av = *(const v16bf*)(wpc + (size_t)j * 512);
          acc[j] = WMMA_BF16(av, bv, acc[j]);
        }
        __syncthreads();
      }
    }
  }

#pragma unroll
  for (int j = 0; j < MT; ++j) {
    const int co0 = (ctb + j) << 4;
    float yv[8];
#pragma unroll
    for (int r = 0; r < 8; ++r) {
      const int ch = co0 + r + khalf;
      float y = acc[j][r];
      if (EPI == EPI_BIAS) y += bias[ch];
      if (EPI == EPI_BN || EPI == EPI_BN_RELU || EPI == EPI_BN_SIGMOID) {
        const float sc = bng[ch] * rsqrtf(bnv[ch] + 1e-5f);
        y = y * sc + (bnb[ch] - bnm[ch] * sc);
      }
      if (EPI == EPI_BN_RELU) y = fmaxf(y, 0.f);
      if (EPI == EPI_BN_SIGMOID) y = 1.f / (1.f + __expf(-y));
      yv[r] = y;
    }

    if (nvalid) {
      if (OPM) {
        // transposed bf16 store: 8 consecutive channels -> contiguous d run
        const int ch0 = co0 + khalf;
        const int g = ch0 >> 8;
        const int hh = (ch0 >> 6) & 3;
        const int d0 = ch0 & 63;
        __bf16* dp = dstb + ((((size_t)b * 2 + g) * 4 + hh) * Nd + ncol) * 64 + d0;
        v4bf p0, p1;
        p0[0] = (__bf16)yv[0]; p0[1] = (__bf16)yv[1];
        p0[2] = (__bf16)yv[2]; p0[3] = (__bf16)yv[3];
        p1[0] = (__bf16)yv[4]; p1[1] = (__bf16)yv[5];
        p1[2] = (__bf16)yv[6]; p1[3] = (__bf16)yv[7];
        *(v4bf*)dp = p0;
        *(v4bf*)(dp + 4) = p1;
      } else if (OBF) {
#pragma unroll
        for (int r = 0; r < 8; ++r)
          dstb[((size_t)b * Cout + co0 + r + khalf) * Nd + ncol] = (__bf16)yv[r];
      } else {
#pragma unroll
        for (int r = 0; r < 8; ++r) {
          const size_t oidx = ((size_t)b * Cout + co0 + r + khalf) * Nd + ncol;
          float y = yv[r];
          if (EPI == EPI_ADDPREV) y += dstf[oidx];
          if (EPI == EPI_ADDPREV_LEAKY) {
            y += dstf[oidx];
            y = (y > 0.f) ? y : 0.05f * y;
          }
          dstf[oidx] = y;
        }
      }
    }
  }
}

// Flash-attention, transposed-score formulation; one wave32 per
// (batch, head, 16-query tile). All operands live in padded buffers
// (Npad columns; col N = extra k/v row, cols N+1.. = zeros), so every
// hot-loop load is a plain unconditional vector load.
__global__ __launch_bounds__(32) void attn_flash(
    const __bf16* __restrict__ qkT, const __bf16* __restrict__ vsrc,
    float* __restrict__ o, int Bn, int N, int Npad, float scale) {
  (void)Bn;
  const int dh = 64, C = 256;
  const int NTq = (N + 15) >> 4;
  int bid = blockIdx.x;
  const int qt = bid % NTq;
  bid /= NTq;
  const int h = bid & 3;
  const int b = bid >> 2;

  const int lane = threadIdx.x;
  const int m = lane & 15;
  const int khalf = (lane & 16) ? 8 : 0;
  const int q0 = qt << 4;

  const __bf16* qT = qkT + (((size_t)b * 2 + 0) * 4 + h) * (size_t)Npad * 64;
  const __bf16* kT = qkT + (((size_t)b * 2 + 1) * 4 + h) * (size_t)Npad * 64;
  const __bf16* vb = vsrc + ((size_t)b * C + h * dh) * Npad;

  const int qi = q0 + m;
  const __bf16* qrow = qT + (size_t)qi * 64;
  const v16bf qf0 = frag16(qrow, khalf);
  const v16bf qf1 = frag16(qrow + 32, khalf);

  float Mq = -1e30f, Lq = 0.f;
  v8f oa0 = {}, oa1 = {}, oa2 = {}, oa3 = {};
  const __bf16 z = (__bf16)0.f;

  const int NTk = (N + 16) >> 4; // ceil((N+1)/16)
  for (int kt = 0; kt < NTk; ++kt) {
    const __bf16* krow = kT + ((size_t)(kt << 4) + m) * 64;
    const v16bf ka0 = frag16(krow, khalf);
    const v16bf ka1 = frag16(krow + 32, khalf);
    v8f st = {};
    st = WMMA_BF16(ka0, qf0, st);
    st = WMMA_BF16(ka1, qf1, st);

    float p8[8];
    float tm = -1e30f;
#pragma unroll
    for (int r = 0; r < 8; ++r) {
      const int krw = (kt << 4) + r + khalf;
      const float sv = (krw <= N) ? st[r] * scale : -1e30f;
      p8[r] = sv;
      tm = fmaxf(tm, sv);
    }
    tm = fmaxf(tm, __shfl_xor(tm, 16, 32));
    const float newM = fmaxf(Mq, tm);
    const float corr = __expf(Mq - newM);
    Mq = newM;
    float ts = 0.f;
    v16bf pb; // P^T directly in B-fragment layout
#pragma unroll
    for (int r = 0; r < 8; ++r) {
      const float p = __expf(p8[r] - newM);
      ts += p;
      pb[r] = (__bf16)p;
      pb[r + 8] = z;
    }
    ts += __shfl_xor(ts, 16, 32);
    Lq = Lq * corr + ts;
#pragma unroll
    for (int r = 0; r < 8; ++r) {
      oa0[r] *= corr;
      oa1[r] *= corr;
      oa2[r] *= corr;
      oa3[r] *= corr;
    }

#pragma unroll
    for (int dt = 0; dt < 4; ++dt) {
      const __bf16* vp = vb + (size_t)(dt * 16 + m) * Npad + (kt << 4) + khalf;
      const v4bf x0 = *(const v4bf*)vp;
      const v4bf x1 = *(const v4bf*)(vp + 4);
      v16bf va;
      va[0] = x0[0]; va[1] = x0[1]; va[2] = x0[2]; va[3] = x0[3];
      va[4] = x1[0]; va[5] = x1[1]; va[6] = x1[2]; va[7] = x1[3];
#pragma unroll
      for (int e = 8; e < 16; ++e) va[e] = z;
      if (dt == 0) oa0 = WMMA_BF16(va, pb, oa0);
      if (dt == 1) oa1 = WMMA_BF16(va, pb, oa1);
      if (dt == 2) oa2 = WMMA_BF16(va, pb, oa2);
      if (dt == 3) oa3 = WMMA_BF16(va, pb, oa3);
    }
  }

  if (qi < N) {
    const float inv = 1.f / Lq;
#pragma unroll
    for (int r = 0; r < 8; ++r) {
      const int d2 = r + khalf;
      o[((size_t)b * C + h * dh + 0 + d2) * N + qi] = oa0[r] * inv;
      o[((size_t)b * C + h * dh + 16 + d2) * N + qi] = oa1[r] * inv;
      o[((size_t)b * C + h * dh + 32 + d2) * N + qi] = oa2[r] * inv;
      o[((size_t)b * C + h * dh + 48 + d2) * N + qi] = oa3[r] * inv;
    }
  }
}

// Per-(b,head,pos) L2 normalization over the 64-channel head dim, in place.
__global__ void headnorm_inplace(float* __restrict__ t, int Bn, int N) {
  const long long total = (long long)Bn * 4 * N;
  long long idx = (long long)blockIdx.x * blockDim.x + threadIdx.x;
  if (idx >= total) return;
  const int n = (int)(idx % N);
  const int h = (int)((idx / N) & 3);
  const int b = (int)(idx / (4LL * N));
  float* base = t + ((size_t)b * 256 + h * 64) * N + n;
  float ss = 0.f;
#pragma unroll 8
  for (int d = 0; d < 64; ++d) {
    const float xv = base[(size_t)d * N];
    ss += xv * xv;
  }
  const float inv = 1.f / fmaxf(sqrtf(ss), 0.001f);
#pragma unroll 8
  for (int d = 0; d < 64; ++d) base[(size_t)d * N] *= inv;
}

// xg = concat(x0, o) * s, in place over s.
__global__ void gate_mul(const float* __restrict__ x0, const float* __restrict__ o,
                         float* __restrict__ s, int Bn, int N) {
  const long long total = (long long)Bn * 512 * N;
  long long idx = (long long)blockIdx.x * blockDim.x + threadIdx.x;
  if (idx >= total) return;
  const int n = (int)(idx % N);
  const int c = (int)((idx / N) % 512);
  const int b = (int)(idx / (512LL * N));
  const float xc = (c < 256) ? x0[((size_t)b * 256 + c) * N + n]
                             : o[((size_t)b * 256 + (c - 256)) * N + n];
  s[idx] = xc * s[idx];
}

// Depthwise 5x5 (pad=2) + BN + ReLU.
__global__ void dw5x5_bn_relu(const float* __restrict__ src, const float* __restrict__ w,
                              const float* __restrict__ g, const float* __restrict__ bb,
                              const float* __restrict__ mm, const float* __restrict__ vv,
                              float* __restrict__ dst, int Bn, int C, int H, int Wd) {
  const int N = H * Wd;
  const long long total = (long long)Bn * C * N;
  long long idx = (long long)blockIdx.x * blockDim.x + threadIdx.x;
  if (idx >= total) return;
  const int n = (int)(idx % N);
  const int c = (int)((idx / N) % C);
  const int b = (int)(idx / ((long long)N * C));
  const int y = n / Wd, x = n % Wd;
  float acc = 0.f;
#pragma unroll
  for (int ky = 0; ky < 5; ++ky) {
    const int yy = y + ky - 2;
#pragma unroll
    for (int kx = 0; kx < 5; ++kx) {
      const int xx = x + kx - 2;
      const bool ib = yy >= 0 && yy < H && xx >= 0 && xx < Wd;
      const int sn = ib ? (yy * Wd + xx) : 0;
      acc += src[((size_t)b * C + c) * N + sn] * (ib ? w[c * 25 + ky * 5 + kx] : 0.f);
    }
  }
  const float sc = g[c] * rsqrtf(vv[c] + 1e-5f);
  const float yv = acc * sc + (bb[c] - mm[c] * sc);
  dst[idx] = fmaxf(yv, 0.f);
}

extern "C" void kernel_launch(void* const* d_in, const int* in_sizes, int n_in,
                              void* d_out, int out_size, void* d_ws, size_t ws_size,
                              hipStream_t stream) {
  (void)in_sizes; (void)n_in; (void)out_size; (void)ws_size;
  const float* x0     = (const float*)d_in[0];
  const float* wqk1_w = (const float*)d_in[1];
  const float* wqk1_b = (const float*)d_in[2];
  const float* wqk2_w = (const float*)d_in[3];
  const float* qk_g   = (const float*)d_in[4];
  const float* qk_b   = (const float*)d_in[5];
  const float* qk_m   = (const float*)d_in[6];
  const float* qk_v   = (const float*)d_in[7];
  const float* wv_w   = (const float*)d_in[8];
  const float* wv_b   = (const float*)d_in[9];
  const float* ek     = (const float*)d_in[10];
  const float* ev     = (const float*)d_in[11];
  const float* se1_w  = (const float*)d_in[12];
  const float* se1_g  = (const float*)d_in[13];
  const float* se1_b  = (const float*)d_in[14];
  const float* se1_m  = (const float*)d_in[15];
  const float* se1_v  = (const float*)d_in[16];
  const float* se2_w  = (const float*)d_in[17];
  const float* se2_g  = (const float*)d_in[18];
  const float* se2_b  = (const float*)d_in[19];
  const float* se2_m  = (const float*)d_in[20];
  const float* se2_v  = (const float*)d_in[21];
  const float* dw1_w  = (const float*)d_in[22];
  const float* dwd_w  = (const float*)d_in[23];
  const float* dw_g   = (const float*)d_in[24];
  const float* dw_b   = (const float*)d_in[25];
  const float* dw_m   = (const float*)d_in[26];
  const float* dw_v   = (const float*)d_in[27];
  const float* dw2_w  = (const float*)d_in[28];
  const float* dw2_b  = (const float*)d_in[29];
  const float* dwsk_w = (const float*)d_in[30];
  const float* fusc_w = (const float*)d_in[31];
  const float* fusc_b = (const float*)d_in[32];
  const float* skip_w = (const float*)d_in[33];

  const int B = 8, H = 38, W = 50, N = H * W; // N = 1900
  const int NT = (N + 15) / 16;               // 119 spatial tiles
  const int Npad = N + 16;

  float* ws = (float*)d_ws;
  const size_t S1 = (size_t)B * 256 * N;
  const size_t QKT_E = (size_t)B * 2 * 4 * Npad * 64; // bf16 elements
  const size_t VB_E  = (size_t)B * 256 * Npad;        // bf16 elements

  float*  t    = ws;                         // S1 : wqk_c1 out / xn
  __bf16* qkT  = (__bf16*)(ws + S1);         // q/k transposed+padded
  __bf16* vbuf = (__bf16*)(ws + S1 + QKT_E / 2);
  float*  obuf = ws + S1 + QKT_E / 2 + VB_E / 2; // S1
  float*  s1b  = obuf + S1;                  // S1/8
  float*  sbuf = s1b + S1 / 8;               // 2*S1 : s then xg
  float*  t2a  = sbuf + 2 * S1;              // 2*S1
  float*  t2b  = t2a + 2 * S1;               // 2*S1
  __bf16* wp   = (__bf16*)(t2b + 2 * S1);    // packed weights

  size_t woff = 0;
  __bf16* wp_qk1 = wp + woff; woff += (size_t)256 * 256;
  __bf16* wp_qk2 = wp + woff; woff += (size_t)512 * 512;
  __bf16* wp_v   = wp + woff; woff += (size_t)256 * 256;
  __bf16* wp_se1 = wp + woff; woff += (size_t)32 * 512 * 9;
  __bf16* wp_se2 = wp + woff; woff += (size_t)512 * 32 * 9;
  __bf16* wp_dw1 = wp + woff; woff += (size_t)512 * 512;
  __bf16* wp_dw2 = wp + woff; woff += (size_t)512 * 512;
  __bf16* wp_dsk = wp + woff; woff += (size_t)512 * 512;
  __bf16* wp_fus = wp + woff; woff += (size_t)256 * 512 * 9;
  __bf16* wp_skp = wp + woff; woff += (size_t)256 * 512;

  auto rp = [&](const float* w, __bf16* d, int Co, int Ci, int KHW) {
    const long long tot = (long long)KHW * Co * Ci;
    repack_w<<<dim3((unsigned)((tot + 255) / 256)), dim3(256), 0, stream>>>(w, d, Co, Ci, KHW);
  };
  rp(wqk1_w, wp_qk1, 256, 256, 1);
  rp(wqk2_w, wp_qk2, 512, 512, 1);
  rp(wv_w,   wp_v,   256, 256, 1);
  rp(se1_w,  wp_se1, 32, 512, 9);
  rp(se2_w,  wp_se2, 512, 32, 9);
  rp(dw1_w,  wp_dw1, 512, 512, 1);
  rp(dw2_w,  wp_dw2, 512, 512, 1);
  rp(dwsk_w, wp_dsk, 512, 512, 1);
  rp(fusc_w, wp_fus, 256, 512, 9);
  rp(skip_w, wp_skp, 256, 512, 1);

  {
    const long long tq = (long long)B * 2 * 4 * (Npad - N) * 64;
    pad_qkT<<<dim3((unsigned)((tq + 255) / 256)), dim3(256), 0, stream>>>(qkT, ek, B, N, Npad);
    const long long tv = (long long)B * 256 * (Npad - N);
    pad_v<<<dim3((unsigned)((tv + 255) / 256)), dim3(256), 0, stream>>>(vbuf, ev, B, N, Npad);
  }

  const float scale = logf((float)(N + 1)) / (logf(1900.0f) * 8.0f);
  dim3 blk256(256), blk64(64), blk32(32);
  const dim3 gridC(B * NT); // every conv: one block per (batch, pos-tile)

  // 1. t = conv1x1(x0) + bias            (Cout=256: 8 waves x MT=2)
  conv_wmma<1, 1, 0, EPI_BIAS, false, 8, 2, false, false><<<gridC, blk256, 0, stream>>>(
      x0, nullptr, wp_qk1, wqk1_b, nullptr, nullptr, nullptr, nullptr, t, nullptr,
      B, 256, 256, H, W, N);
  // 2. xn = normalize heads of t (in place)
  {
    const long long tot = (long long)B * 4 * N;
    headnorm_inplace<<<dim3((unsigned)((tot + 255) / 256)), dim3(256), 0, stream>>>(t, B, N);
  }
  // 3. qk = bn(conv1x1(concat(x0, xn))) -> transposed bf16 (8 waves x MT=4)
  conv_wmma<1, 1, 0, EPI_BN, true, 8, 4, true, true><<<gridC, blk256, 0, stream>>>(
      x0, t, wp_qk2, nullptr, qk_g, qk_b, qk_m, qk_v, nullptr, qkT, B, 512, 512, H, W, Npad);
  // 4. v = conv1x1(x0) + bias -> bf16 [ch][Npad]
  conv_wmma<1, 1, 0, EPI_BIAS, false, 8, 2, true, false><<<gridC, blk256, 0, stream>>>(
      x0, nullptr, wp_v, wv_b, nullptr, nullptr, nullptr, nullptr, nullptr, vbuf,
      B, 256, 256, H, W, Npad);
  // 5. o = attention(q, k+extra_k, v+extra_v)
  attn_flash<<<dim3(B * 4 * NT), blk32, 0, stream>>>(qkT, vbuf, obuf, B, N, Npad, scale);
  // 6. s1 = relu(bn(conv3x3(concat(x0, o))))   512 -> 32 (2 waves x MT=1)
  conv_wmma<3, 3, 1, EPI_BN_RELU, true, 2, 1, false, false><<<gridC, blk64, 0, stream>>>(
      x0, obuf, wp_se1, nullptr, se1_g, se1_b, se1_m, se1_v, s1b, nullptr, B, 512, 32, H, W, N);
  // 7. s = sigmoid(bn(conv3x3(s1)))            32 -> 512
  conv_wmma<3, 3, 1, EPI_BN_SIGMOID, false, 8, 4, false, false><<<gridC, blk256, 0, stream>>>(
      s1b, nullptr, wp_se2, nullptr, se2_g, se2_b, se2_m, se2_v, sbuf, nullptr,
      B, 32, 512, H, W, N);
  // 8. xg = concat(x0, o) * s (in place)
  {
    const long long tot = (long long)B * 512 * N;
    gate_mul<<<dim3((unsigned)((tot + 255) / 256)), dim3(256), 0, stream>>>(x0, obuf, sbuf, B, N);
  }
  // 9. t2a = conv1x1(xg)     (dw1, no bias)
  conv_wmma<1, 1, 0, EPI_NONE, false, 8, 4, false, false><<<gridC, blk256, 0, stream>>>(
      sbuf, nullptr, wp_dw1, nullptr, nullptr, nullptr, nullptr, nullptr, t2a, nullptr,
      B, 512, 512, H, W, N);
  // 10. t2b = relu(bn(depthwise5x5(t2a)))
  {
    const long long tot = (long long)B * 512 * N;
    dw5x5_bn_relu<<<dim3((unsigned)((tot + 255) / 256)), dim3(256), 0, stream>>>(
        t2a, dwd_w, dw_g, dw_b, dw_m, dw_v, t2b, B, 512, H, W);
  }
  // 11. t2a = conv1x1(t2b) + dw2_b
  conv_wmma<1, 1, 0, EPI_BIAS, false, 8, 4, false, false><<<gridC, blk256, 0, stream>>>(
      t2b, nullptr, wp_dw2, dw2_b, nullptr, nullptr, nullptr, nullptr, t2a, nullptr,
      B, 512, 512, H, W, N);
  // 12. t2a = leaky_relu(t2a + conv1x1(xg))    (dwskip)
  conv_wmma<1, 1, 0, EPI_ADDPREV_LEAKY, false, 8, 4, false, false><<<gridC, blk256, 0, stream>>>(
      sbuf, nullptr, wp_dsk, nullptr, nullptr, nullptr, nullptr, nullptr, t2a, nullptr,
      B, 512, 512, H, W, N);
  // 13. out = conv3x3(t2a) + fusc_b            512 -> 256
  conv_wmma<3, 3, 1, EPI_BIAS, false, 8, 2, false, false><<<gridC, blk256, 0, stream>>>(
      t2a, nullptr, wp_fus, fusc_b, nullptr, nullptr, nullptr, nullptr, (float*)d_out, nullptr,
      B, 512, 256, H, W, N);
  // 14. out += conv1x1(concat(x0, o))          (skip)
  conv_wmma<1, 1, 0, EPI_ADDPREV, true, 8, 2, false, false><<<gridC, blk256, 0, stream>>>(
      x0, obuf, wp_skp, nullptr, nullptr, nullptr, nullptr, nullptr, (float*)d_out, nullptr,
      B, 512, 256, H, W, N);
}
